// GNNProteinDTI_81441169867011
// MI455X (gfx1250) — compile-verified, matrix operand order checked by reading, not run on hardware
//
#include <hip/hip_runtime.h>
#include <hip/hip_bf16.h>
#include <math.h>
#include <stdint.h>

typedef __attribute__((ext_vector_type(16))) __bf16 v16bf;
typedef __attribute__((ext_vector_type(8)))  __bf16 v8bf;
typedef __attribute__((ext_vector_type(8)))  float  v8f;

#define DEV_INLINE __device__ __forceinline__

DEV_INLINE float eluf(float x)  { return x > 0.f ? x : (__expf(x) - 1.f); }
DEV_INLINE float reluf(float x) { return x > 0.f ? x : 0.f; }
DEV_INLINE float actf(float x, int act) { return act == 1 ? reluf(x) : (act == 2 ? eluf(x) : x); }

DEV_INLINE float atomicMaxF(float* addr, float val) {
  int old = __float_as_int(*addr);
  while (true) {
    float f = __int_as_float(old);
    if (f >= val) break;
    int prev = atomicCAS((int*)addr, old, __float_as_int(val));
    if (prev == old) break;
    old = prev;
  }
  return __int_as_float(old);
}

// ---- CDNA5 async copy: 16 bytes global -> LDS per active lane (ASYNCcnt) ----
DEV_INLINE void async_cp16(unsigned lds_off, const void* gptr) {
  asm volatile("global_load_async_to_lds_b128 %0, %1, off"
               :: "v"(lds_off), "v"((unsigned long long)(uintptr_t)gptr)
               : "memory");
}
DEV_INLINE void wait_async0() { asm volatile("s_wait_asynccnt 0x0" ::: "memory"); }

// WMMA bf16 A/B fragment from an LDS row (layout per CDNA5 ISA 7.12.2):
// lane L: row = L&15, khalf = L>>4; elems 0..7 = K[8*khalf ..), 8..15 = K[16+8*khalf ..).
DEV_INLINE v16bf frag16(const __bf16* p) {
  v8bf lo = *(const v8bf*)p;
  v8bf hi = *(const v8bf*)(p + 16);
  return __builtin_shufflevector(lo, hi, 0,1,2,3,4,5,6,7,8,9,10,11,12,13,14,15);
}

// ---------------------------------------------------------------------------
// Tiled WMMA GEMM, bf16 in, f32 accumulate: C = act(A[M,K] @ BT[N,K]^T + bias)
// Block 128x128, 8 waves (2x4), wave tile 64x32. Double-buffered LDS with
// async-to-LDS copies overlapping WMMA compute. Requires M%128==0, K%32==0.
// ---------------------------------------------------------------------------
__global__ __launch_bounds__(256)
void gemm_bf16(const __bf16* __restrict__ A, int lda,
               const __bf16* __restrict__ BT,    // [N][K] row-major (pre-transposed)
               const float* __restrict__ bias,   // [N] or nullptr
               void* __restrict__ Cv, int ldc,
               int M, int N, int K, int act, int out_bf16) {
  __shared__ __attribute__((aligned(16))) __bf16 As[2][128 * 40];
  __shared__ __attribute__((aligned(16))) __bf16 Bs[2][128 * 40];
  const int tid = threadIdx.x;
  const int lane = tid & 31, wave = tid >> 5;
  const int wm = wave >> 2, wn = wave & 3;
  const int m0 = blockIdx.x * 128, n0 = blockIdx.y * 128;
  const int mlane = lane & 15, khalf = lane >> 4;
  v8bf zero8;
  for (int i = 0; i < 8; i++) zero8[i] = (__bf16)0.f;

  v8f acc[4][2];
  for (int i = 0; i < 4; i++)
    for (int j = 0; j < 2; j++)
      for (int r = 0; r < 8; r++) acc[i][j][r] = 0.f;

  auto stage = [&](int k0, int buf) {
    unsigned ab = (unsigned)(uintptr_t)&As[buf][0];
    unsigned bb = (unsigned)(uintptr_t)&Bs[buf][0];
    for (int i = tid; i < 512; i += 256) {      // A: 128 rows x 4 chunks of 16B
      int r = i >> 2, c4 = i & 3;
      async_cp16(ab + (unsigned)((r * 40 + c4 * 8) * 2),
                 A + (long)(m0 + r) * lda + k0 + c4 * 8);
    }
    for (int i = tid; i < 512; i += 256) {      // B rows of BT; zero-fill n >= N
      int r = i >> 2, c4 = i & 3;
      int gn = n0 + r;
      if (gn < N)
        async_cp16(bb + (unsigned)((r * 40 + c4 * 8) * 2),
                   BT + (long)gn * K + k0 + c4 * 8);
      else
        *(v8bf*)&Bs[buf][r * 40 + c4 * 8] = zero8;
    }
  };

  const int nt = K >> 5;
  stage(0, 0);
  for (int it = 0; it < nt; it++) {
    const int buf = it & 1;
    wait_async0();        // this wave's copies for tile `it` are in LDS
    __syncthreads();      // publish across waves; prior tile's readers done
    if (it + 1 < nt) stage((it + 1) * 32, buf ^ 1);  // overlap with compute

    v16bf af[4], bfr[2];
    for (int mi = 0; mi < 4; mi++)
      af[mi] = frag16(&As[buf][(wm * 64 + mi * 16 + mlane) * 40 + khalf * 8]);
    for (int ni = 0; ni < 2; ni++)
      bfr[ni] = frag16(&Bs[buf][(wn * 32 + ni * 16 + mlane) * 40 + khalf * 8]);
    for (int mi = 0; mi < 4; mi++)
      for (int ni = 0; ni < 2; ni++)
        acc[mi][ni] = __builtin_amdgcn_wmma_f32_16x16x32_bf16(
            false, af[mi], false, bfr[ni], (short)0, acc[mi][ni], false, false);
  }

  for (int mi = 0; mi < 4; mi++)
    for (int ni = 0; ni < 2; ni++) {
      int n = n0 + wn * 32 + ni * 16 + mlane;
      int mb = m0 + wm * 64 + mi * 16 + khalf * 8;
      if (n < N) {
        float bv = bias ? bias[n] : 0.f;
        for (int r = 0; r < 8; r++) {
          int m = mb + r;
          if (m < M) {
            float v = actf(acc[mi][ni][r] + bv, act);
            if (out_bf16) ((__bf16*)Cv)[(long)m * ldc + n] = (__bf16)v;
            else          ((float*)Cv)[(long)m * ldc + n] = v;
          }
        }
      }
    }
}

// ---------------------------------------------------------------------------
// Implicit-GEMM conv1d (channels-last bf16, Cin=64), double-buffered async:
// Out[(b,l), co] = elu( sum_{t,ci} X[(b, l+t-pad), ci] * WT[co][t*64+ci] + bias[co] )
// ---------------------------------------------------------------------------
__global__ __launch_bounds__(256)
void conv1d_bf16(const __bf16* __restrict__ X,    // [Bb*L, 64]
                 const __bf16* __restrict__ WT,   // [N][KT*64]
                 const float* __restrict__ bias,  // [N]
                 __bf16* __restrict__ Out,        // [Bb*L, N]
                 int Bb, int L, int KT, int pad, int N, int act) {
  __shared__ __attribute__((aligned(16))) __bf16 As[2][128 * 40];
  __shared__ __attribute__((aligned(16))) __bf16 Bs[2][128 * 40];
  const int Ktot = KT * 64;
  const int M = Bb * L;
  const int tid = threadIdx.x;
  const int lane = tid & 31, wave = tid >> 5;
  const int wm = wave >> 2, wn = wave & 3;
  const int m0 = blockIdx.x * 128, n0 = blockIdx.y * 128;
  const int mlane = lane & 15, khalf = lane >> 4;
  v8bf zero8;
  for (int i = 0; i < 8; i++) zero8[i] = (__bf16)0.f;

  v8f acc[4][2];
  for (int i = 0; i < 4; i++)
    for (int j = 0; j < 2; j++)
      for (int r = 0; r < 8; r++) acc[i][j][r] = 0.f;

  auto stage = [&](int k0, int buf) {
    unsigned ab = (unsigned)(uintptr_t)&As[buf][0];
    unsigned bb = (unsigned)(uintptr_t)&Bs[buf][0];
    // A: patch gather; each 16B chunk = 8 contiguous channels of row (b, l+t-pad)
    for (int i = tid; i < 512; i += 256) {
      int r = i >> 2, c4 = i & 3;
      int gm = m0 + r;
      int b = gm / L, l = gm - b * L;
      int k = k0 + c4 * 8;
      int t = k >> 6, ci = k & 63;
      int li = l + t - pad;
      unsigned loff = ab + (unsigned)((r * 40 + c4 * 8) * 2);
      if (li >= 0 && li < L)
        async_cp16(loff, X + ((long)b * L + li) * 64 + ci);
      else
        *(v8bf*)&As[buf][r * 40 + c4 * 8] = zero8;
    }
    for (int i = tid; i < 512; i += 256) {
      int r = i >> 2, c4 = i & 3;
      int gn = n0 + r;
      if (gn < N)
        async_cp16(bb + (unsigned)((r * 40 + c4 * 8) * 2),
                   WT + (long)gn * Ktot + k0 + c4 * 8);
      else
        *(v8bf*)&Bs[buf][r * 40 + c4 * 8] = zero8;
    }
  };

  const int nt = Ktot >> 5;
  stage(0, 0);
  for (int it = 0; it < nt; it++) {
    const int buf = it & 1;
    wait_async0();
    __syncthreads();
    if (it + 1 < nt) stage((it + 1) * 32, buf ^ 1);

    v16bf af[4], bfr[2];
    for (int mi = 0; mi < 4; mi++)
      af[mi] = frag16(&As[buf][(wm * 64 + mi * 16 + mlane) * 40 + khalf * 8]);
    for (int ni = 0; ni < 2; ni++)
      bfr[ni] = frag16(&Bs[buf][(wn * 32 + ni * 16 + mlane) * 40 + khalf * 8]);
    for (int mi = 0; mi < 4; mi++)
      for (int ni = 0; ni < 2; ni++)
        acc[mi][ni] = __builtin_amdgcn_wmma_f32_16x16x32_bf16(
            false, af[mi], false, bfr[ni], (short)0, acc[mi][ni], false, false);
  }

  for (int mi = 0; mi < 4; mi++)
    for (int ni = 0; ni < 2; ni++) {
      int n = n0 + wn * 32 + ni * 16 + mlane;
      int mb = m0 + wm * 64 + mi * 16 + khalf * 8;
      if (n < N) {
        float bv = bias[n];
        for (int r = 0; r < 8; r++) {
          int m = mb + r;
          if (m < M) Out[(long)m * N + n] = (__bf16)actf(acc[mi][ni][r] + bv, act);
        }
      }
    }
}

// ---------------------------------------------------------------------------
// Small scalar kernels
// ---------------------------------------------------------------------------
__global__ void fill_f32(float* p, float v, int n) {
  int i = blockIdx.x * blockDim.x + threadIdx.x;
  if (i < n) p[i] = v;
}

__global__ void embed_k(const int* __restrict__ seq, const float* __restrict__ Pe,
                        __bf16* __restrict__ pe, long total) {  // total = B*L*64
  long i = (long)blockIdx.x * blockDim.x + threadIdx.x;
  if (i >= total) return;
  long pos = i >> 6; int c = (int)(i & 63);
  pe[i] = (__bf16)Pe[seq[pos] * 64 + c];
}

// Generic dense weight transpose+convert: WT[n][k] = W[k][n]
__global__ void pack_wT_k(const float* __restrict__ W, __bf16* __restrict__ WT,
                          int K, int N) {
  int i = blockIdx.x * blockDim.x + threadIdx.x;
  if (i >= N * K) return;
  int n = i / K, k = i - n * K;
  WT[i] = (__bf16)W[k * N + n];
}

__global__ void pack_k1T_k(const float* __restrict__ K1, __bf16* __restrict__ WT) {
  int i = blockIdx.x * blockDim.x + threadIdx.x;  // 64 * 192
  if (i >= 64 * 192) return;
  int co = i / 192, k = i - co * 192;
  int t = k >> 6, ci = k & 63;
  WT[i] = (__bf16)K1[(co * 64 + ci) * 3 + t];
}

__global__ void pack_k2T_k(const float* __restrict__ K2, __bf16* __restrict__ WT) {
  int i = blockIdx.x * blockDim.x + threadIdx.x;  // 128 * 320
  if (i >= 128 * 320) return;
  int co = i / 320, k = i - co * 320;
  int t = k >> 6, ci = k & 63;
  WT[i] = (__bf16)K2[(co * 64 + ci) * 5 + t];
}

__global__ void gat_proj5_k(const float* __restrict__ x, const float* __restrict__ W1,
                            float* __restrict__ h, int Nn) {
  int i = blockIdx.x * blockDim.x + threadIdx.x;
  if (i >= Nn * 128) return;
  int n = i >> 7, c = i & 127;
  float s = 0.f;
  for (int k = 0; k < 5; k++) s += x[n * 5 + k] * W1[k * 128 + c];
  h[i] = s;
}

__global__ void att_scores_k(const float* __restrict__ h, const float* __restrict__ asw,
                             const float* __restrict__ adw, float* __restrict__ as_,
                             float* __restrict__ ad_, int Nn, int H, int C) {
  int i = blockIdx.x * blockDim.x + threadIdx.x;
  if (i >= Nn * H) return;
  int n = i / H, hd = i % H;
  const float* hp = h + (long)n * H * C + hd * C;
  float s = 0.f, d = 0.f;
  for (int c = 0; c < C; c++) { float v = hp[c]; s += v * asw[hd * C + c]; d += v * adw[hd * C + c]; }
  as_[i] = s; ad_[i] = d;
}

DEV_INLINE void edge_sd(const int* ei, int E, int e, int& s, int& d) {
  if (e < E) { s = ei[e]; d = ei[E + e]; } else { s = d = e - E; }
}
DEV_INLINE float lrelu02(float x) { return x > 0.f ? x : 0.2f * x; }

__global__ void edge_max_k(const int* __restrict__ ei, int E, int Nn,
                           const float* __restrict__ as_, const float* __restrict__ ad_,
                           float* __restrict__ mx, int H) {
  int i = blockIdx.x * blockDim.x + threadIdx.x;
  int tot = (E + Nn) * H;
  if (i >= tot) return;
  int e = i / H, hd = i % H, s, d;
  edge_sd(ei, E, e, s, d);
  atomicMaxF(&mx[d * H + hd], lrelu02(as_[s * H + hd] + ad_[d * H + hd]));
}

__global__ void edge_den_k(const int* __restrict__ ei, int E, int Nn,
                           const float* __restrict__ as_, const float* __restrict__ ad_,
                           const float* __restrict__ mx, float* __restrict__ den, int H) {
  int i = blockIdx.x * blockDim.x + threadIdx.x;
  int tot = (E + Nn) * H;
  if (i >= tot) return;
  int e = i / H, hd = i % H, s, d;
  edge_sd(ei, E, e, s, d);
  float sc = lrelu02(as_[s * H + hd] + ad_[d * H + hd]);
  atomicAdd(&den[d * H + hd], __expf(sc - mx[d * H + hd]));
}

__global__ void edge_scatter_k(const int* __restrict__ ei, int E, int Nn,
                               const float* __restrict__ as_, const float* __restrict__ ad_,
                               const float* __restrict__ mx, const float* __restrict__ den,
                               const float* __restrict__ h, float* __restrict__ out,
                               int H, int C) {
  int i = blockIdx.x * blockDim.x + threadIdx.x;
  int tot = (E + Nn) * H;
  if (i >= tot) return;
  int e = i / H, hd = i % H, s, d;
  edge_sd(ei, E, e, s, d);
  float sc = lrelu02(as_[s * H + hd] + ad_[d * H + hd]);
  float w = __expf(sc - mx[d * H + hd]);
  float alpha = w / (den[d * H + hd] + 1e-16f);
  const float* hs = h + (long)s * H * C + hd * C;
  float* od = out + (long)d * H * C + hd * C;
  for (int c = 0; c < C; c++) atomicAdd(&od[c], hs[c] * alpha);
}

// g := elu(g + bias); optional bf16 mirror for feeding WMMA GEMMs
__global__ void bias_elu_k(float* __restrict__ g, const float* __restrict__ b,
                           __bf16* __restrict__ ob, int Nn, int Ch) {
  int i = blockIdx.x * blockDim.x + threadIdx.x;
  if (i >= Nn * Ch) return;
  float v = eluf(g[i] + b[i % Ch]);
  g[i] = v;
  if (ob) ob[i] = (__bf16)v;
}

__global__ void seg_pool_k(const float* __restrict__ g, const int* __restrict__ batch,
                           float* __restrict__ sums, float* __restrict__ cnt, int Nn) {
  int i = blockIdx.x * blockDim.x + threadIdx.x;
  if (i >= Nn * 128) return;
  int n = i >> 7, c = i & 127;
  int b = batch[n];
  atomicAdd(&sums[b * 128 + c], g[i]);
  if (c == 0) atomicAdd(&cnt[b], 1.f);
}

__global__ void mean_k(const float* __restrict__ sums, const float* __restrict__ cnt,
                       __bf16* __restrict__ o, int Gg) {
  int i = blockIdx.x * blockDim.x + threadIdx.x;
  if (i >= Gg * 128) return;
  o[i] = (__bf16)(sums[i] / fmaxf(cnt[i >> 7], 1.f));
}

__global__ void maxpool_k(const __bf16* __restrict__ p, __bf16* __restrict__ o,
                          int Bb, int L) {
  int i = blockIdx.x * blockDim.x + threadIdx.x;
  if (i >= Bb * 128) return;
  int b = i >> 7, c = i & 127;
  float m = -1e30f;
  for (int l = 0; l < L; l++) m = fmaxf(m, (float)p[((long)b * L + l) * 128 + c]);
  o[i] = (__bf16)m;
}

__global__ void final_k(const float* __restrict__ h, const float* __restrict__ Wo,
                        const float* __restrict__ bo, float* __restrict__ out, int Bb) {
  int b = blockIdx.x * blockDim.x + threadIdx.x;
  if (b >= Bb) return;
  float s = bo[0];
  for (int c = 0; c < 64; c++) s += h[b * 64 + c] * Wo[c];
  out[b] = s;
}

// ---------------------------------------------------------------------------
extern "C" void kernel_launch(void* const* d_in, const int* in_sizes, int n_in,
                              void* d_out, int out_size, void* d_ws, size_t ws_size,
                              hipStream_t stream) {
  const int Nn = 24576, E = 98304, Gg = 512, Bb = 512, L = 1000;

  const float* x    = (const float*)d_in[0];
  const int*   ei   = (const int*)d_in[1];
  const int*   batch= (const int*)d_in[2];
  const int*   seq  = (const int*)d_in[3];
  const float* W1   = (const float*)d_in[4];
  const float* asw1 = (const float*)d_in[5];
  const float* adw1 = (const float*)d_in[6];
  const float* b1   = (const float*)d_in[7];
  const float* W2   = (const float*)d_in[8];
  const float* asw2 = (const float*)d_in[9];
  const float* adw2 = (const float*)d_in[10];
  const float* b2   = (const float*)d_in[11];
  const float* Wd   = (const float*)d_in[12];
  const float* bd   = (const float*)d_in[13];
  const float* Pe   = (const float*)d_in[14];
  const float* K1   = (const float*)d_in[15];
  const float* bk1  = (const float*)d_in[16];
  const float* K2   = (const float*)d_in[17];
  const float* bk2  = (const float*)d_in[18];
  const float* Wp   = (const float*)d_in[19];
  const float* bp   = (const float*)d_in[20];
  const float* Wf1  = (const float*)d_in[21];
  const float* bf1  = (const float*)d_in[22];
  const float* Wf2  = (const float*)d_in[23];
  const float* bf2  = (const float*)d_in[24];
  const float* Wo   = (const float*)d_in[25];
  const float* bo   = (const float*)d_in[26];
  float* out = (float*)d_out;

  char* w = (char*)d_ws;
  size_t off = 0;
  auto allocB = [&](size_t bytes) -> char* {
    char* p = w + off;
    off = (off + bytes + 255) & ~(size_t)255;
    return p;
  };
  float* h1     = (float*)allocB((size_t)Nn * 128 * 4);
  float* as1    = (float*)allocB((size_t)Nn * 2 * 4);
  float* ad1    = (float*)allocB((size_t)Nn * 2 * 4);
  float* mx1    = (float*)allocB((size_t)Nn * 2 * 4);
  float* den1   = (float*)allocB((size_t)Nn * 2 * 4);
  float* g1     = (float*)allocB((size_t)Nn * 128 * 4);
  __bf16* g1b   = (__bf16*)allocB((size_t)Nn * 128 * 2);
  float* h2     = (float*)allocB((size_t)Nn * 128 * 4);
  float* as2    = (float*)allocB((size_t)Nn * 4);
  float* ad2    = (float*)allocB((size_t)Nn * 4);
  float* mx2    = (float*)allocB((size_t)Nn * 4);
  float* den2   = (float*)allocB((size_t)Nn * 4);
  float* g2     = (float*)allocB((size_t)Nn * 128 * 4);
  float* sums   = (float*)allocB((size_t)Gg * 128 * 4);
  float* cnt    = (float*)allocB((size_t)Gg * 4);
  __bf16* drugb = (__bf16*)allocB((size_t)Gg * 128 * 2);
  __bf16* W2T   = (__bf16*)allocB((size_t)128 * 128 * 2);
  __bf16* WdT   = (__bf16*)allocB((size_t)256 * 128 * 2);
  __bf16* WpT   = (__bf16*)allocB((size_t)256 * 128 * 2);
  __bf16* Wf1T  = (__bf16*)allocB((size_t)128 * 512 * 2);
  __bf16* Wf2T  = (__bf16*)allocB((size_t)64 * 128 * 2);
  __bf16* W1T   = (__bf16*)allocB((size_t)64 * 192 * 2);
  __bf16* W2cT  = (__bf16*)allocB((size_t)128 * 320 * 2);
  __bf16* pe    = (__bf16*)allocB((size_t)Bb * L * 64 * 2);
  __bf16* p1    = (__bf16*)allocB((size_t)Bb * L * 64 * 2);
  __bf16* p2    = (__bf16*)allocB((size_t)Bb * L * 128 * 2);
  __bf16* protb = (__bf16*)allocB((size_t)Bb * 128 * 2);
  __bf16* hcatb = (__bf16*)allocB((size_t)Bb * 512 * 2);
  __bf16* hf1b  = (__bf16*)allocB((size_t)Bb * 128 * 2);
  float* hf2    = (float*)allocB((size_t)Bb * 64 * 4);

  auto g1d = [](long n) { return dim3((unsigned)((n + 255) / 256)); };
  const dim3 blk(256);

  // ---- weight packs (bf16, transposed) ----
  pack_wT_k<<<g1d(128 * 128), blk, 0, stream>>>(W2, W2T, 128, 128);
  pack_wT_k<<<g1d(256 * 128), blk, 0, stream>>>(Wd, WdT, 128, 256);
  pack_wT_k<<<g1d(256 * 128), blk, 0, stream>>>(Wp, WpT, 128, 256);
  pack_wT_k<<<g1d(128 * 512), blk, 0, stream>>>(Wf1, Wf1T, 512, 128);
  pack_wT_k<<<g1d(64 * 128), blk, 0, stream>>>(Wf2, Wf2T, 128, 64);
  pack_k1T_k<<<g1d(64 * 192), blk, 0, stream>>>(K1, W1T);
  pack_k2T_k<<<g1d(128 * 320), blk, 0, stream>>>(K2, W2cT);

  // ---- drug branch: GAT layer 1 (H=2, C=64) ----
  hipMemsetAsync(g1, 0, (size_t)Nn * 128 * 4, stream);
  hipMemsetAsync(den1, 0, (size_t)Nn * 2 * 4, stream);
  fill_f32<<<g1d(Nn * 2), blk, 0, stream>>>(mx1, -1e30f, Nn * 2);
  gat_proj5_k<<<g1d((long)Nn * 128), blk, 0, stream>>>(x, W1, h1, Nn);
  att_scores_k<<<g1d(Nn * 2), blk, 0, stream>>>(h1, asw1, adw1, as1, ad1, Nn, 2, 64);
  long et1 = (long)(E + Nn) * 2;
  edge_max_k<<<g1d(et1), blk, 0, stream>>>(ei, E, Nn, as1, ad1, mx1, 2);
  edge_den_k<<<g1d(et1), blk, 0, stream>>>(ei, E, Nn, as1, ad1, mx1, den1, 2);
  edge_scatter_k<<<g1d(et1), blk, 0, stream>>>(ei, E, Nn, as1, ad1, mx1, den1, h1, g1, 2, 64);
  bias_elu_k<<<g1d((long)Nn * 128), blk, 0, stream>>>(g1, b1, g1b, Nn, 128);

  // ---- GAT layer 2 projection via WMMA GEMM ----
  gemm_bf16<<<dim3(Nn / 128, 1), blk, 0, stream>>>(g1b, 128, W2T, nullptr, h2, 128,
                                                   Nn, 128, 128, 0, 0);
  hipMemsetAsync(g2, 0, (size_t)Nn * 128 * 4, stream);
  hipMemsetAsync(den2, 0, (size_t)Nn * 4, stream);
  fill_f32<<<g1d(Nn), blk, 0, stream>>>(mx2, -1e30f, Nn);
  att_scores_k<<<g1d(Nn), blk, 0, stream>>>(h2, asw2, adw2, as2, ad2, Nn, 1, 128);
  long et2 = (long)(E + Nn);
  edge_max_k<<<g1d(et2), blk, 0, stream>>>(ei, E, Nn, as2, ad2, mx2, 1);
  edge_den_k<<<g1d(et2), blk, 0, stream>>>(ei, E, Nn, as2, ad2, mx2, den2, 1);
  edge_scatter_k<<<g1d(et2), blk, 0, stream>>>(ei, E, Nn, as2, ad2, mx2, den2, h2, g2, 1, 128);
  bias_elu_k<<<g1d((long)Nn * 128), blk, 0, stream>>>(g2, b2, (__bf16*)nullptr, Nn, 128);

  // ---- global mean pool + drug dense (relu) -> hcat[:, :256] ----
  hipMemsetAsync(sums, 0, (size_t)Gg * 128 * 4, stream);
  hipMemsetAsync(cnt, 0, (size_t)Gg * 4, stream);
  seg_pool_k<<<g1d((long)Nn * 128), blk, 0, stream>>>(g2, batch, sums, cnt, Nn);
  mean_k<<<g1d(Gg * 128), blk, 0, stream>>>(sums, cnt, drugb, Gg);
  gemm_bf16<<<dim3(Gg / 128, 2), blk, 0, stream>>>(drugb, 128, WdT, bd, hcatb, 512,
                                                   Gg, 256, 128, 1, 1);

  // ---- protein branch ----
  embed_k<<<g1d((long)Bb * L * 64), blk, 0, stream>>>(seq, Pe, pe, (long)Bb * L * 64);
  conv1d_bf16<<<dim3(Bb * L / 128, 1), blk, 0, stream>>>(pe, W1T, bk1, p1, Bb, L, 3, 1, 64, 2);
  conv1d_bf16<<<dim3(Bb * L / 128, 1), blk, 0, stream>>>(p1, W2cT, bk2, p2, Bb, L, 5, 2, 128, 2);
  maxpool_k<<<g1d(Bb * 128), blk, 0, stream>>>(p2, protb, Bb, L);
  gemm_bf16<<<dim3(Bb / 128, 2), blk, 0, stream>>>(protb, 128, WpT, bp, hcatb + 256, 512,
                                                   Bb, 256, 128, 1, 1);

  // ---- head ----
  gemm_bf16<<<dim3(Bb / 128, 1), blk, 0, stream>>>(hcatb, 512, Wf1T, bf1, hf1b, 128,
                                                   Bb, 128, 512, 1, 1);
  gemm_bf16<<<dim3(Bb / 128, 1), blk, 0, stream>>>(hf1b, 128, Wf2T, bf2, hf2, 64,
                                                   Bb, 64, 128, 1, 0);
  final_k<<<g1d(Bb), blk, 0, stream>>>(hf2, Wo, bo, out, Bb);
}
